// Mamba2MIL_49340584296559
// MI455X (gfx1250) — matrix operand order repaired and checked by reading, not compiled, not run on hardware
//
#include <hip/hip_runtime.h>
#include <stdint.h>

// ---------------------------------------------------------------------------
// Types for CDNA5 WMMA (wave32): D(f32 16x16) = A(bf16 16x32) x B(bf16 32x16) + C
// ---------------------------------------------------------------------------
typedef __attribute__((ext_vector_type(16))) __bf16 v16bf;
typedef __attribute__((ext_vector_type(8)))  __bf16 v8bf;
typedef __attribute__((ext_vector_type(8)))  float  v8f;

#define DEVINL static __device__ __forceinline__

// ---------------------------------------------------------------------------
// Async global -> LDS (CDNA5 GLOBAL_LOAD_ASYNC_TO_LDS_B128); probe-verified:
// builtin exists with int4* parameters. Fallback keeps file compilable.
// ---------------------------------------------------------------------------
#if defined(__AMDGCN__) && __has_builtin(__builtin_amdgcn_global_load_async_to_lds_b128)
#define HAS_ASYNC 1
typedef int v4i_async __attribute__((vector_size(16)));
#else
#define HAS_ASYNC 0
#endif

DEVINL void cp16(void* l, const void* g) {
#if HAS_ASYNC
  __builtin_amdgcn_global_load_async_to_lds_b128((v4i_async*)g, (v4i_async*)l, 0, 0);
#else
  *(float4*)l = *(const float4*)g;
#endif
}

DEVINL void wait_async() {
#if HAS_ASYNC
#if __has_builtin(__builtin_amdgcn_s_wait_asynccnt)
  __builtin_amdgcn_s_wait_asynccnt(0);
#else
  asm volatile("s_wait_asynccnt 0" ::: "memory");
#endif
#endif
}

DEVINL v8f wmma_bf16(v16bf a, v16bf b, v8f c) {
  return __builtin_amdgcn_wmma_f32_16x16x32_bf16(false, a, false, b, (short)0, c,
                                                 false, false);
}

// ---------------------------------------------------------------------------
// Fragment builders (ISA 7.12.2 wave32 layouts): two ds_load_b128 each.
// A 16x32: lanes 0-15 -> M=lane, halves 0..7 = K0..7, 8..15 = K16..23
//          lanes16-31 -> M=lane-16, halves 0..7 = K8..15, 8..15 = K24..31
// B 32x16: lanes 0-15 -> N=lane, halves = K0..7,K16..23; lanes16-31 -> K8..15,K24..31
// C/D 16x16: lanes 0-15 -> N=lane, vgpr j = M=j; lanes16-31 -> M=j+8
// ---------------------------------------------------------------------------
DEVINL v16bf ldfrag_a(const __bf16* __restrict__ src, int stride, int m0, int k0,
                      int lane) {
  int half = lane >> 4, r = lane & 15;
  const __bf16* row = src + (m0 + r) * stride + k0 + (half << 3);
  v8bf lo = *(const v8bf*)row;
  v8bf hi = *(const v8bf*)(row + 16);
  return __builtin_shufflevector(lo, hi, 0, 1, 2, 3, 4, 5, 6, 7, 8, 9, 10, 11, 12,
                                 13, 14, 15);
}

// B fragment from n-major (transposed) storage: B[k][n] = src[n*stride + k]
DEVINL v16bf ldfrag_bT(const __bf16* __restrict__ src, int stride, int k0, int n0,
                       int lane) {
  int half = lane >> 4, r = lane & 15;
  const __bf16* row = src + (n0 + r) * stride + k0 + (half << 3);
  v8bf lo = *(const v8bf*)row;
  v8bf hi = *(const v8bf*)(row + 16);
  return __builtin_shufflevector(lo, hi, 0, 1, 2, 3, 4, 5, 6, 7, 8, 9, 10, 11, 12,
                                 13, 14, 15);
}

DEVINL float block_reduce_sum256(float v, float* sred, int tid) {
  sred[tid] = v;
  __syncthreads();
#pragma unroll
  for (int s = 128; s > 0; s >>= 1) {
    if (tid < s) sred[tid] += sred[tid + s];
    __syncthreads();
  }
  float r = sred[0];
  __syncthreads();
  return r;
}

// ---------------------------------------------------------------------------
// Model dimensions
// ---------------------------------------------------------------------------
#define SEQ    4096
#define NBATCH 2
#define NROWS  8192
#define IN_DIM 1024
#define DM     512
#define DI     2048
#define NDS    256
#define NH     32
#define HD     64
#define CONVD  2560
#define DIP    4640
#define NCLS   4

// ---------------------------------------------------------------------------
// f32 -> bf16 conversion (weights once per launch; activations by producers)
// ---------------------------------------------------------------------------
__global__ __launch_bounds__(256) void cvt_bf16_kernel(const float* __restrict__ src,
                                                       __bf16* __restrict__ dst,
                                                       int n) {
  int i = (blockIdx.x * 256 + threadIdx.x) * 4;
  if (i >= n) return;
  float4 v = *(const float4*)(src + i);
  dst[i] = (__bf16)v.x;
  dst[i + 1] = (__bf16)v.y;
  dst[i + 2] = (__bf16)v.z;
  dst[i + 3] = (__bf16)v.w;
}

// ---------------------------------------------------------------------------
// GEMM: C(MxN) = act(A(MxK)bf16 @ W(NxK)^T bf16 + bias). 128 thr (4 waves),
// 64x64 C tile, DOUBLE-BUFFERED async global->LDS tile staging: tile k+1 is
// in flight (ASYNCcnt) while WMMAs consume tile k.
// ---------------------------------------------------------------------------
template <int ACT, bool HASBIAS>  // ACT: 0 none, 1 relu, 2 tanh
__global__ __launch_bounds__(128) void gemm_bias_act(
    const __bf16* __restrict__ A, const __bf16* __restrict__ W,
    const float* __restrict__ bias, float* __restrict__ C, int M, int N, int K) {
  __shared__ alignas(16) __bf16 As[2][64 * 40];
  __shared__ alignas(16) __bf16 Ws[2][64 * 40];
  const int tid = threadIdx.x;
  const int lane = tid & 31, wave = tid >> 5;
  const int mb = blockIdx.y * 64, nb = blockIdx.x * 64;
  v8f acc[4] = {};

  auto stage = [&](int bufi, int k0) {
#pragma unroll
    for (int i = 0; i < 2; ++i) {
      int cch = tid + (i << 7);      // 256 16B-chunks per 64x32 bf16 tile
      int rr = cch >> 2, q = cch & 3;
      cp16(&As[bufi][rr * 40 + q * 8], A + (size_t)(mb + rr) * K + k0 + q * 8);
      int nn = nb + rr;
      if (nn >= N) nn = N - 1;  // clamp: duplicated rows feed discarded cols
      cp16(&Ws[bufi][rr * 40 + q * 8], W + (size_t)nn * K + k0 + q * 8);
    }
  };

  stage(0, 0);
  int cur = 0;
  for (int k0 = 0; k0 < K; k0 += 32) {
    wait_async();    // this wave's copies (incl. buffer `cur`) complete
    __syncthreads(); // all waves' copies visible; prior readers of cur^1 done
    if (k0 + 32 < K) stage(cur ^ 1, k0 + 32);  // issue next tile, no wait
    v16bf af = ldfrag_a(As[cur], 40, wave * 16, 0, lane);
#pragma unroll
    for (int nt = 0; nt < 4; ++nt) {
      v16bf bf_ = ldfrag_bT(Ws[cur], 40, 0, nt * 16, lane);
      acc[nt] = wmma_bf16(af, bf_, acc[nt]);
    }
    cur ^= 1;
  }
  const int half = lane >> 4, r = lane & 15;
#pragma unroll
  for (int nt = 0; nt < 4; ++nt) {
    int col = nb + nt * 16 + r;
    if (col >= N) continue;
    float bv = HASBIAS ? bias[col] : 0.f;
#pragma unroll
    for (int j = 0; j < 8; ++j) {
      int row = mb + wave * 16 + j + half * 8;
      float v = acc[nt][j] + bv;
      if (ACT == 1) v = fmaxf(v, 0.f);
      if (ACT == 2) v = tanhf(v);
      C[(size_t)row * N + col] = v;
    }
  }
}

// ---------------------------------------------------------------------------
// LayerNorm over DM=512 per row; emits f32 and bf16 copies
// ---------------------------------------------------------------------------
__global__ __launch_bounds__(256) void ln_kernel(const float* __restrict__ X,
                                                 const float* __restrict__ w,
                                                 const float* __restrict__ bv,
                                                 float* __restrict__ O,
                                                 __bf16* __restrict__ Ob) {
  __shared__ float sred[256];
  const int tid = threadIdx.x;
  const size_t row = blockIdx.x;
  float v0 = X[row * 512 + tid], v1 = X[row * 512 + 256 + tid];
  float mean = block_reduce_sum256(v0 + v1, sred, tid) * (1.f / 512.f);
  float d0 = v0 - mean, d1 = v1 - mean;
  float var = block_reduce_sum256(d0 * d0 + d1 * d1, sred, tid) * (1.f / 512.f);
  float rs = rsqrtf(var + 1e-5f);
  float o0 = d0 * rs * w[tid] + bv[tid];
  float o1 = d1 * rs * w[tid + 256] + bv[tid + 256];
  O[row * 512 + tid] = o0;
  O[row * 512 + 256 + tid] = o1;
  Ob[row * 512 + tid] = (__bf16)o0;
  Ob[row * 512 + 256 + tid] = (__bf16)o1;
}

// ---------------------------------------------------------------------------
// dt = softplus(zx[...,4608+h] + dt_bias); a = dt * (-exp(A_log))
// ---------------------------------------------------------------------------
__global__ __launch_bounds__(256) void dt_kernel(const float* __restrict__ zx,
                                                 const float* __restrict__ dt_bias,
                                                 const float* __restrict__ A_log,
                                                 float* __restrict__ dtv,
                                                 float* __restrict__ av) {
  int idx = blockIdx.x * 256 + threadIdx.x;
  int hh = idx & 31;
  size_t row = (size_t)(idx >> 5);
  float raw = zx[row * DIP + 4608 + hh] + dt_bias[hh];
  float dt = (raw > 20.f) ? raw : log1pf(__expf(raw));
  dtv[idx] = dt;
  av[idx] = dt * (-__expf(A_log[hh]));
}

// ---------------------------------------------------------------------------
// Causal depthwise conv(4) + bias + SiLU; split into x(f32), x*dt(bf16), B, C
// ---------------------------------------------------------------------------
__global__ __launch_bounds__(256) void conv_silu_split(
    const float* __restrict__ zx, const float* __restrict__ convw,
    const float* __restrict__ convb, const float* __restrict__ dtv,
    float* __restrict__ Xf, __bf16* __restrict__ Xdt, __bf16* __restrict__ Bo,
    __bf16* __restrict__ Co) {
  int c = blockIdx.x * 256 + threadIdx.x;
  int l = blockIdx.y, b = blockIdx.z;
  size_t row = (size_t)(b * SEQ + l);
  float acc = convb[c];
  const float* wv = convw + c * 4;
#pragma unroll
  for (int j = 0; j < 4; ++j) {
    int ls = l - 3 + j;
    if (ls >= 0) acc += zx[(size_t)(b * SEQ + ls) * DIP + DI + c] * wv[j];
  }
  float sv = acc / (1.f + __expf(-acc));
  if (c < DI) {
    Xf[row * DI + c] = sv;
    float dt = dtv[row * NH + (c >> 6)];
    Xdt[row * DI + c] = (__bf16)(sv * dt);
  } else if (c < DI + NDS) {
    Bo[row * NDS + (c - DI)] = (__bf16)sv;
  } else {
    Co[row * NDS + (c - DI - NDS)] = (__bf16)sv;
  }
}

// ---------------------------------------------------------------------------
// SSD chunked scan. Block per (head,batch), 256 thr (8 waves), 64 chunks.
// State S (64p x 256n f32) in registers; bf16 shadow in LDS for Yo B-operand.
// Fused pass: each 32-wide async C/B panel feeds BOTH the CB (intra-chunk)
// and Yo (state read-out) WMMAs from one A-fragment. exp(acs) row scale is
// applied once at the Y epilogue; dec(l) scale folded into PbT staging.
// ---------------------------------------------------------------------------
#define SST 264  // bf16 state shadow stride (528B rows, 16B aligned)
__global__ __launch_bounds__(256) void ssd_kernel(
    const __bf16* __restrict__ Xdt, const __bf16* __restrict__ Bg,
    const __bf16* __restrict__ Cg, const float* __restrict__ ag,
    float* __restrict__ Yg) {
  __shared__ alignas(16) __bf16 S16[64 * SST];  // state shadow [p][n]
  __shared__ alignas(16) __bf16 XsT[64 * 72];   // x*dt transposed [p][l]
  __shared__ alignas(16) __bf16 Ms[64 * 72];    // masked CB [l][s]
  __shared__ alignas(16) __bf16 Pc[64 * 40];    // C panel [l][n-panel]
  __shared__ alignas(16) __bf16 PbU[64 * 40];   // B panel [l][n] / dec*B [n][l]
  __shared__ float acs[64], els[64], decs[64];
  __shared__ float expTsh;

  const int h = blockIdx.x, b = blockIdx.y;
  const int tid = threadIdx.x, lane = tid & 31, wave = tid >> 5;
  const int half = lane >> 4, r = lane & 15;
  const int spt = wave >> 1;   // state tile p-index owned by this wave
  const int sodd = wave & 1;   // state tile n-parity owned by this wave

  v8f Sreg[8] = {};  // state tiles: slot s -> (pt=spt, nt=2*s+sodd)

  const uint32_t* Bu0 = (const uint32_t*)Bg;
  const uint32_t* Xu0 = (const uint32_t*)Xdt;

  for (int c = 0; c < 64; ++c) {
    const size_t lbase = (size_t)(b * SEQ + c * 64);
    __syncthreads();

    // dump register state -> bf16 shadow
#pragma unroll
    for (int s = 0; s < 8; ++s) {
      int nt = (s << 1) | sodd;
#pragma unroll
      for (int j = 0; j < 8; ++j) {
        int p = spt * 16 + j + half * 8, n = nt * 16 + r;
        S16[p * SST + n] = (__bf16)Sreg[s][j];
      }
    }
    // load x*dt chunk transposed: XsT[p][l]
    {
      size_t xbase = (lbase * NH + h) * (HD / 2);  // uints
#pragma unroll
      for (int i = 0; i < 8; ++i) {
        int idx = tid + (i << 8);
        int l = idx >> 5, pp = (idx & 31) * 2;
        uint32_t wv = Xu0[xbase + (size_t)l * (NH * HD / 2) + (idx & 31)];
        XsT[pp * 72 + l] = ((const __bf16*)&wv)[0];
        XsT[(pp + 1) * 72 + l] = ((const __bf16*)&wv)[1];
      }
    }
    if (tid < 64) acs[tid] = ag[(lbase + tid) * NH + h];
    __syncthreads();
    if (tid == 0) {
      float s = 0.f;
      for (int l = 0; l < 64; ++l) { s += acs[l]; acs[l] = s; }
      expTsh = __expf(acs[63]);
    }
    __syncthreads();
    if (tid < 64) {
      els[tid] = __expf(acs[tid]);
      decs[tid] = __expf(acs[63] - acs[tid]);
    }

    const __bf16* Crow = Cg + lbase * NDS;
    const __bf16* Brow = Bg + lbase * NDS;

    // ---- Phase 1 (fused): CB[l][s] += C.B^T ; Yo[l][p] += C.S^T ----
    v8f cbacc[2] = {}, yo[2] = {};
#pragma unroll
    for (int pan = 0; pan < 8; ++pan) {
      __syncthreads();
      {
        int l = tid >> 2, q = tid & 3;  // 256 x 16B per 64x32 panel
        cp16(&Pc[l * 40 + q * 8], Crow + (size_t)l * NDS + pan * 32 + q * 8);
        cp16(&PbU[l * 40 + q * 8], Brow + (size_t)l * NDS + pan * 32 + q * 8);
      }
      wait_async();
      __syncthreads();
#pragma unroll
      for (int tt = 0; tt < 2; ++tt) {
        int tau = wave * 2 + tt, lt = tau >> 2, nt3 = tau & 3;
        v16bf af = ldfrag_a(Pc, 40, lt * 16, 0, lane);  // shared A for both
        cbacc[tt] = wmma_bf16(af, ldfrag_bT(PbU, 40, 0, nt3 * 16, lane), cbacc[tt]);
        yo[tt] = wmma_bf16(af, ldfrag_bT(S16, SST, pan * 32, nt3 * 16, lane), yo[tt]);
      }
    }
    __syncthreads();
#pragma unroll
    for (int tt = 0; tt < 2; ++tt) {
      int tau = wave * 2 + tt, lt = tau >> 2, st = tau & 3;
#pragma unroll
      for (int j = 0; j < 8; ++j) {
        int l = lt * 16 + j + half * 8, s = st * 16 + r;
        float v = (s <= l) ? cbacc[tt][j] * __expf(acs[l] - acs[s]) : 0.f;
        Ms[l * 72 + s] = (__bf16)v;
      }
    }
    __syncthreads();

    // ---- Phase 2: Y = Ms @ X + exp(acs[l]) * Yo ----
#pragma unroll
    for (int tt = 0; tt < 2; ++tt) {
      int tau = wave * 2 + tt, lt = tau >> 2, pt = tau & 3;
      v8f yd = {};
#pragma unroll
      for (int kb = 0; kb < 64; kb += 32) {
        v16bf af = ldfrag_a(Ms, 72, lt * 16, kb, lane);
        v16bf bf_ = ldfrag_bT(XsT, 72, kb, pt * 16, lane);
        yd = wmma_bf16(af, bf_, yd);
      }
#pragma unroll
      for (int j = 0; j < 8; ++j) {
        int l = lt * 16 + j + half * 8, p = pt * 16 + r;
        Yg[((lbase + l) * NH + h) * HD + p] = yd[j] + els[l] * yo[tt][j];
      }
    }

    // ---- Phase 3: S = exp(T)*S + X^T @ (dec*B) ----
#pragma unroll
    for (int s = 0; s < 8; ++s) Sreg[s] = Sreg[s] * expTsh;
    __bf16* PbT = PbU;  // reuse panel buffer as dec-scaled B^T [n][l], stride 72
#pragma unroll
    for (int pan = 0; pan < 8; ++pan) {
      __syncthreads();
#pragma unroll
      for (int i = 0; i < 4; ++i) {
        int idx = tid + (i << 8);
        int l = idx >> 4, nn = (idx & 15) * 2;
        uint32_t wv = Bu0[lbase * (NDS / 2) + (size_t)l * 128 + pan * 16 + (idx & 15)];
        float d = decs[l];
        PbT[nn * 72 + l] = (__bf16)(d * (float)((const __bf16*)&wv)[0]);
        PbT[(nn + 1) * 72 + l] = (__bf16)(d * (float)((const __bf16*)&wv)[1]);
      }
      __syncthreads();
#pragma unroll
      for (int kb = 0; kb < 64; kb += 32) {
        v16bf af = ldfrag_a(XsT, 72, spt * 16, kb, lane);
        v16bf bf_ = ldfrag_bT(PbT, 72, kb, sodd * 16, lane);
        Sreg[pan] = wmma_bf16(af, bf_, Sreg[pan]);
      }
    }
  }
}

// ---------------------------------------------------------------------------
// y = (yssd + Dv*x) * silu(z); RMS-norm over DI; * gn_w; bf16 out
// ---------------------------------------------------------------------------
__global__ __launch_bounds__(256) void gate_norm_kernel(
    const float* __restrict__ Y, const float* __restrict__ Xf,
    const float* __restrict__ zx, const float* __restrict__ Dv,
    const float* __restrict__ gnw, __bf16* __restrict__ Yn) {
  __shared__ float sred[256];
  const int tid = threadIdx.x;
  const size_t row = blockIdx.x;
  float vals[8], ss = 0.f;
#pragma unroll
  for (int i = 0; i < 8; ++i) {
    int d = tid + (i << 8);
    float y = Y[row * DI + d] + Dv[d >> 6] * Xf[row * DI + d];
    float z = zx[row * DIP + d];
    y *= z / (1.f + __expf(-z));
    vals[i] = y;
    ss += y * y;
  }
  float tot = block_reduce_sum256(ss, sred, tid);
  float rs = rsqrtf(tot * (1.f / (float)DI) + 1e-5f);
#pragma unroll
  for (int i = 0; i < 8; ++i) {
    int d = tid + (i << 8);
    Yn[row * DI + d] = (__bf16)(vals[i] * rs * gnw[d]);
  }
}

__global__ __launch_bounds__(256) void resid_mask_kernel(float* __restrict__ h,
                                                         const float* __restrict__ tmp,
                                                         const float* __restrict__ mask) {
  size_t i = (size_t)blockIdx.x * 256 + threadIdx.x;
  size_t row = i >> 9;
  h[i] = (h[i] + tmp[i]) * mask[row];
}

__global__ __launch_bounds__(128) void score_kernel(const float* __restrict__ T,
                                                    const float* __restrict__ w2,
                                                    const float* __restrict__ b2,
                                                    const float* __restrict__ mask,
                                                    float* __restrict__ sc) {
  __shared__ float sred[128];
  const int tid = threadIdx.x;
  const size_t row = blockIdx.x;
  sred[tid] = T[row * 128 + tid] * w2[tid];
  __syncthreads();
#pragma unroll
  for (int s = 64; s > 0; s >>= 1) {
    if (tid < s) sred[tid] += sred[tid + s];
    __syncthreads();
  }
  if (tid == 0) sc[row] = (mask[row] == 0.f) ? -1e9f : (sred[0] + b2[0]);
}

__global__ __launch_bounds__(256) void softmax_kernel(const float* __restrict__ sc,
                                                      float* __restrict__ att) {
  __shared__ float sred[256];
  const int b = blockIdx.x, tid = threadIdx.x;
  float v[16], mx = -1e30f;
#pragma unroll
  for (int i = 0; i < 16; ++i) {
    v[i] = sc[b * SEQ + tid + (i << 8)];
    mx = fmaxf(mx, v[i]);
  }
  sred[tid] = mx;
  __syncthreads();
#pragma unroll
  for (int s = 128; s > 0; s >>= 1) {
    if (tid < s) sred[tid] = fmaxf(sred[tid], sred[tid + s]);
    __syncthreads();
  }
  float m = sred[0];
  __syncthreads();
  float ss = 0.f;
#pragma unroll
  for (int i = 0; i < 16; ++i) {
    v[i] = __expf(v[i] - m);
    ss += v[i];
  }
  float inv = 1.f / block_reduce_sum256(ss, sred, tid);
#pragma unroll
  for (int i = 0; i < 16; ++i) att[b * SEQ + tid + (i << 8)] = v[i] * inv;
}

__global__ __launch_bounds__(256) void pool_kernel(const float* __restrict__ att,
                                                   const float* __restrict__ hn,
                                                   float* __restrict__ pooled) {
  const int b = blockIdx.y;
  const int d = blockIdx.x * 256 + threadIdx.x;
  float acc = 0.f;
  for (int l = 0; l < SEQ; ++l)
    acc += att[b * SEQ + l] * hn[((size_t)b * SEQ + l) * DM + d];
  pooled[b * DM + d] = acc;
}

__global__ void head_kernel(const float* __restrict__ pooled,
                            const float* __restrict__ clsw,
                            const float* __restrict__ clsb, float* __restrict__ out) {
  __shared__ float lg[8];
  const int tid = threadIdx.x;
  if (tid < 8) {
    int b = tid >> 2, cc = tid & 3;
    float acc = clsb[cc];
    for (int d = 0; d < DM; ++d) acc += pooled[b * DM + d] * clsw[cc * DM + d];
    lg[tid] = acc;
    out[tid] = acc;  // logits
  }
  __syncthreads();
  if (tid < 2) {
    int b = tid;
    float mx = lg[b * 4];
    int am = 0;
    for (int c2 = 1; c2 < 4; ++c2)
      if (lg[b * 4 + c2] > mx) { mx = lg[b * 4 + c2]; am = c2; }
    float e[4], s = 0.f;
    for (int c2 = 0; c2 < 4; ++c2) { e[c2] = __expf(lg[b * 4 + c2] - mx); s += e[c2]; }
    for (int c2 = 0; c2 < 4; ++c2) out[8 + b * 4 + c2] = e[c2] / s;  // Y_prob
    out[16 + b] = (float)am;                                          // Y_hat
  }
}

// ---------------------------------------------------------------------------
// Host launch
// ---------------------------------------------------------------------------
extern "C" void kernel_launch(void* const* d_in, const int* in_sizes, int n_in,
                              void* d_out, int out_size, void* d_ws, size_t ws_size,
                              hipStream_t stream) {
  (void)in_sizes; (void)n_in; (void)out_size; (void)ws_size;
  const float* x       = (const float*)d_in[0];
  const float* mask    = (const float*)d_in[1];
  const float* fc1_w   = (const float*)d_in[2];
  const float* fc1_b   = (const float*)d_in[3];
  const float* ln_w    = (const float*)d_in[4];
  const float* ln_b    = (const float*)d_in[5];
  const float* in_w    = (const float*)d_in[6];
  const float* conv_w  = (const float*)d_in[7];
  const float* conv_b  = (const float*)d_in[8];
  const float* dt_bias = (const float*)d_in[9];
  const float* A_log   = (const float*)d_in[10];
  const float* Dv      = (const float*)d_in[11];
  const float* gn_w    = (const float*)d_in[12];
  const float* out_w   = (const float*)d_in[13];
  const float* norm_w  = (const float*)d_in[14];
  const float* norm_b  = (const float*)d_in[15];
  const float* attn_w1 = (const float*)d_in[16];
  const float* attn_b1 = (const float*)d_in[17];
  const float* attn_w2 = (const float*)d_in[18];
  const float* attn_b2 = (const float*)d_in[19];
  const float* cls_w   = (const float*)d_in[20];
  const float* cls_b   = (const float*)d_in[21];
  float* out = (float*)d_out;

  // workspace carve-out (f32 region then bf16 region; all 16B aligned)
  float* wsf = (float*)d_ws;
  size_t off = 0;
  auto allocF = [&](size_t n) { float* p = wsf + off; off += n; return p; };
  float* h    = allocF((size_t)NROWS * DM);
  float* hn   = allocF((size_t)NROWS * DM);
  float* zx   = allocF((size_t)NROWS * DIP);
  float* Xf   = allocF((size_t)NROWS * DI);
  float* dtv  = allocF((size_t)NROWS * NH);
  float* av   = allocF((size_t)NROWS * NH);
  float* yssd = allocF((size_t)NROWS * DI);
  float* tmp  = allocF((size_t)NROWS * DM);
  float* tbuf = allocF((size_t)NROWS * 128);
  float* sc   = allocF((size_t)NROWS);
  __bf16* wsb = (__bf16*)(wsf + off);
  size_t offb = 0;
  auto allocB = [&](size_t n) { __bf16* p = wsb + offb; offb += n; return p; };
  __bf16* Xdt   = allocB((size_t)NROWS * DI);
  __bf16* Bo    = allocB((size_t)NROWS * NDS);
  __bf16* Co    = allocB((size_t)NROWS * NDS);
  __bf16* x_bf  = allocB((size_t)NROWS * IN_DIM);
  __bf16* hn_bf = allocB((size_t)NROWS * DM);
  __bf16* yn_bf = allocB((size_t)NROWS * DI);
  __bf16* wfc1  = allocB((size_t)DM * IN_DIM);
  __bf16* win   = allocB((size_t)2 * DIP * DM);
  __bf16* wout  = allocB((size_t)2 * DM * DI);
  __bf16* wattn = allocB((size_t)128 * DM);

  float* attO = out + 18;     // Att (2,4096)
  float* poolO = out + 8210;  // pooled (2,512)

  auto cvt = [&](const float* s, __bf16* d, size_t n) {
    cvt_bf16_kernel<<<(unsigned)((n + 1023) / 1024), 256, 0, stream>>>(s, d, (int)n);
  };
  cvt(x, x_bf, (size_t)NROWS * IN_DIM);
  cvt(fc1_w, wfc1, (size_t)DM * IN_DIM);
  cvt(in_w, win, (size_t)2 * DIP * DM);
  cvt(out_w, wout, (size_t)2 * DM * DI);
  cvt(attn_w1, wattn, (size_t)128 * DM);

  // fc1 + relu
  gemm_bias_act<1, true><<<dim3(DM / 64, NROWS / 64), 128, 0, stream>>>(
      x_bf, wfc1, fc1_b, h, NROWS, DM, IN_DIM);

  for (int i = 0; i < 2; ++i) {
    ln_kernel<<<NROWS, 256, 0, stream>>>(h, ln_w + i * DM, ln_b + i * DM, hn, hn_bf);
    gemm_bias_act<0, false><<<dim3((DIP + 63) / 64, NROWS / 64), 128, 0, stream>>>(
        hn_bf, win + (size_t)i * DIP * DM, nullptr, zx, NROWS, DIP, DM);
    dt_kernel<<<(NROWS * NH) / 256, 256, 0, stream>>>(zx, dt_bias + i * NH,
                                                      A_log + i * NH, dtv, av);
    conv_silu_split<<<dim3(CONVD / 256, SEQ, NBATCH), 256, 0, stream>>>(
        zx, conv_w + (size_t)i * CONVD * 4, conv_b + (size_t)i * CONVD, dtv, Xf, Xdt,
        Bo, Co);
    ssd_kernel<<<dim3(NH, NBATCH), 256, 0, stream>>>(Xdt, Bo, Co, av, yssd);
    gate_norm_kernel<<<NROWS, 256, 0, stream>>>(yssd, Xf, zx, Dv + i * NH,
                                                gn_w + (size_t)i * DI, yn_bf);
    gemm_bias_act<0, false><<<dim3(DM / 64, NROWS / 64), 128, 0, stream>>>(
        yn_bf, wout + (size_t)i * DM * DI, nullptr, tmp, NROWS, DM, DI);
    resid_mask_kernel<<<(NROWS * DM) / 256, 256, 0, stream>>>(h, tmp, mask);
  }

  ln_kernel<<<NROWS, 256, 0, stream>>>(h, norm_w, norm_b, hn, hn_bf);
  gemm_bias_act<2, true><<<dim3(128 / 64, NROWS / 64), 128, 0, stream>>>(
      hn_bf, wattn, attn_b1, tbuf, NROWS, 128, DM);
  score_kernel<<<NROWS, 128, 0, stream>>>(tbuf, attn_w2, attn_b2, mask, sc);
  softmax_kernel<<<NBATCH, 256, 0, stream>>>(sc, attO);
  pool_kernel<<<dim3(DM / 256, NBATCH), 256, 0, stream>>>(attO, hn, poolO);
  head_kernel<<<1, 32, 0, stream>>>(poolO, cls_w, cls_b, out);
}
